// Elman_33397665694132
// MI455X (gfx1250) — matrix operand order repaired and checked by reading, not compiled, not run on hardware
//
#include <hip/hip_runtime.h>
#include <hip/hip_bf16.h>

// ---- problem constants (from reference) ----
#define T_STEPS 512
#define IN_DIM  512
#define BATCH   256
#define HID     1024
#define WID     512
#define KH      (HID + IN_DIM)   // 1536
#define KT_H    (KH / 32)        // 48 k-tiles for the h-GEMM (32 from h, 16 from x)
#define KT_HH   (HID / 32)       // 32 k-tiles of the h part
#define KT_HX   (IN_DIM / 32)    // 16 k-tiles of the x part
#define KT_Y    (HID / 32)       // 32 k-tiles for the y-GEMM

typedef __attribute__((ext_vector_type(16))) __bf16 v16bf;
typedef __attribute__((ext_vector_type(8)))  __bf16 v8bf;
typedef __attribute__((ext_vector_type(8)))  float  v8f;

__device__ __forceinline__ __bf16 f2bf(float f) { return (__bf16)f; }

__device__ __forceinline__ v8f wmma_bf16(v16bf a, v16bf b, v8f c) {
    return __builtin_amdgcn_wmma_f32_16x16x32_bf16(false, a, false, b,
                                                   (short)0, c, false, false);
}

// ---------------------------------------------------------------------------
// Pack W (fp32 row-major [M][K]) into bf16 WMMA A-fragment order:
// element ((mt*KT + kt)*32 + lane)*16 + j holds
//   A row m = mt*16 + (lane&15)
//   A col k = kt*32 + (lane<16 ? 0 : 16) + j
// (contiguous k order, matching the B-fragment loads).
// ---------------------------------------------------------------------------
__global__ void pack_wh(const float* __restrict__ W, __bf16* __restrict__ Wp) {
    int idx = blockIdx.x * blockDim.x + threadIdx.x;   // < 64*KT_H*32*16
    int j  = idx & 15;
    int l  = (idx >> 4) & 31;
    int t2 = idx >> 9;
    int kt = t2 % KT_H;
    int mt = t2 / KT_H;
    int m  = mt * 16 + (l & 15);
    int k  = kt * 32 + ((l < 16) ? 0 : 16) + j;
    Wp[idx] = f2bf(W[(size_t)m * KH + k]);
}

__global__ void pack_wy(const float* __restrict__ W, __bf16* __restrict__ Wp) {
    int idx = blockIdx.x * blockDim.x + threadIdx.x;   // < 32*KT_Y*32*16
    int j  = idx & 15;
    int l  = (idx >> 4) & 31;
    int t2 = idx >> 9;
    int kt = t2 & (KT_Y - 1);
    int mt = t2 >> 5;
    int m  = mt * 16 + (l & 15);
    int k  = kt * 32 + ((l < 16) ? 0 : 16) + j;
    Wp[idx] = f2bf(W[(size_t)m * HID + k]);
}

// hT_init[n][m] = h0[m]  (transposed, broadcast across batch), bf16
__global__ void init_h(const float* __restrict__ h0, __bf16* __restrict__ hT) {
    int idx = blockIdx.x * blockDim.x + threadIdx.x;   // < BATCH*HID
    int m = idx & (HID - 1);
    hT[idx] = f2bf(h0[m]);
}

// transpose + convert one timestep of x: xT[n][k] = bf16(x_t[k][n])
__global__ void xpose(const float* __restrict__ xt, __bf16* __restrict__ xT) {
    int idx = blockIdx.x * blockDim.x + threadIdx.x;   // < IN_DIM*BATCH
    int n = idx & (BATCH - 1);
    int k = idx >> 8;
    xT[(size_t)n * IN_DIM + k] = f2bf(xt[(size_t)k * BATCH + n]);
}

// ---------------------------------------------------------------------------
// h-step: h_newT = sigmoid(W_h @ [h; x_t] + b_h)^T
// grid = (8, 8), block = 128 (4 wave32). Wave computes a 2x2 set of 16x16
// tiles: mtiles {mt0, mt0+32} x ntiles {nt0, nt0+8}. Both A fragments are
// reused across both B fragments -> 8 b128 loads per 4 WMMAs.
// All operands are direct contiguous 32-byte global loads (no LDS/sync).
// ---------------------------------------------------------------------------
__global__ void __launch_bounds__(128) elman_hstep(
    const __bf16* __restrict__ Wp,      // packed A fragments [64][48][32][16]
    const __bf16* __restrict__ hT,      // [BATCH][HID] bf16 (transposed h)
    const __bf16* __restrict__ xT,      // [BATCH][IN] bf16 (transposed x_t)
    const float*  __restrict__ bh,      // [HID]
    __bf16*       __restrict__ hTn)     // [BATCH][HID] bf16 out (transposed)
{
    const int lane = threadIdx.x & 31;
    const int wave = threadIdx.x >> 5;
    const int mt0  = blockIdx.y * 4 + wave;    // 0..31
    const int mt1  = mt0 + 32;                 // 32..63
    const int nb0  = blockIdx.x * 16;          // batch cols 0..127
    const int nb1  = nb0 + 128;                // batch cols 128..255
    const int n    = lane & 15;
    const int kb   = (lane < 16) ? 0 : 16;     // B-fragment K base

    v8f c00 = {}, c01 = {}, c10 = {}, c11 = {};
    const __bf16* a0 = Wp + ((size_t)mt0 * KT_H) * 512 + (size_t)lane * 16;
    const __bf16* a1 = Wp + ((size_t)mt1 * KT_H) * 512 + (size_t)lane * 16;
    const __bf16* bh0 = hT + (size_t)(nb0 + n) * HID + kb;
    const __bf16* bh1 = hT + (size_t)(nb1 + n) * HID + kb;
    const __bf16* bx0 = xT + (size_t)(nb0 + n) * IN_DIM + kb;
    const __bf16* bx1 = xT + (size_t)(nb1 + n) * IN_DIM + kb;

    // phase 1: K tiles 0..31 (recurrent h part)
    #pragma unroll 4
    for (int kt = 0; kt < KT_HH; ++kt) {
        __builtin_prefetch(a0 + (size_t)(kt + 8) * 512, 0, 3);
        __builtin_prefetch(a1 + (size_t)(kt + 8) * 512, 0, 3);
        v16bf af0 = *(const v16bf*)(a0 + (size_t)kt * 512);
        v16bf af1 = *(const v16bf*)(a1 + (size_t)kt * 512);
        v16bf b0  = *(const v16bf*)(bh0 + kt * 32);
        v16bf b1  = *(const v16bf*)(bh1 + kt * 32);
        c00 = wmma_bf16(af0, b0, c00);
        c01 = wmma_bf16(af0, b1, c01);
        c10 = wmma_bf16(af1, b0, c10);
        c11 = wmma_bf16(af1, b1, c11);
    }
    // phase 2: K tiles 32..47 (input x part)
    #pragma unroll 4
    for (int kt = 0; kt < KT_HX; ++kt) {
        v16bf af0 = *(const v16bf*)(a0 + (size_t)(KT_HH + kt) * 512);
        v16bf af1 = *(const v16bf*)(a1 + (size_t)(KT_HH + kt) * 512);
        v16bf b0  = *(const v16bf*)(bx0 + kt * 32);
        v16bf b1  = *(const v16bf*)(bx1 + kt * 32);
        c00 = wmma_bf16(af0, b0, c00);
        c01 = wmma_bf16(af0, b1, c01);
        c10 = wmma_bf16(af1, b0, c10);
        c11 = wmma_bf16(af1, b1, c11);
    }

    // epilogue: bias + sigmoid; C row r -> M = r + (lane<16?0:8), so each lane
    // owns 8 contiguous M columns of hT_next[n][*]: one 16B store per tile.
    const int half = (lane < 16) ? 0 : 8;
    const int mb0  = mt0 * 16 + half;
    const int mb1  = mt1 * 16 + half;
    v8f bias0 = *(const v8f*)(bh + mb0);
    v8f bias1 = *(const v8f*)(bh + mb1);
    v8bf p00, p01, p10, p11;
    #pragma unroll
    for (int r = 0; r < 8; ++r) {
        p00[r] = f2bf(1.0f / (1.0f + __expf(-(c00[r] + bias0[r]))));
        p01[r] = f2bf(1.0f / (1.0f + __expf(-(c01[r] + bias0[r]))));
        p10[r] = f2bf(1.0f / (1.0f + __expf(-(c10[r] + bias1[r]))));
        p11[r] = f2bf(1.0f / (1.0f + __expf(-(c11[r] + bias1[r]))));
    }
    *(v8bf*)(hTn + (size_t)(nb0 + n) * HID + mb0) = p00;
    *(v8bf*)(hTn + (size_t)(nb1 + n) * HID + mb0) = p01;
    *(v8bf*)(hTn + (size_t)(nb0 + n) * HID + mb1) = p10;
    *(v8bf*)(hTn + (size_t)(nb1 + n) * HID + mb1) = p11;
}

// ---------------------------------------------------------------------------
// y-step: y = sigmoid(W_y @ h_new + b_y), fp32 out, reference layout [WID][BATCH]
// grid = (8, 4), block = 128 (4 wave32); wave computes 2x2 tiles.
// ---------------------------------------------------------------------------
__global__ void __launch_bounds__(128) elman_ystep(
    const __bf16* __restrict__ Wyp,     // packed A fragments [32][32][32][16]
    const __bf16* __restrict__ hT,      // [BATCH][HID] bf16 (transposed h_new)
    const float*  __restrict__ by,      // [WID]
    float*        __restrict__ out)     // [WID][BATCH] fp32 (this timestep)
{
    const int lane = threadIdx.x & 31;
    const int wave = threadIdx.x >> 5;
    const int mt0  = blockIdx.y * 4 + wave;    // 0..15
    const int mt1  = mt0 + 16;                 // 16..31
    const int nb0  = blockIdx.x * 16;
    const int nb1  = nb0 + 128;
    const int n    = lane & 15;
    const int kb   = (lane < 16) ? 0 : 16;

    v8f c00 = {}, c01 = {}, c10 = {}, c11 = {};
    const __bf16* a0 = Wyp + ((size_t)mt0 * KT_Y) * 512 + (size_t)lane * 16;
    const __bf16* a1 = Wyp + ((size_t)mt1 * KT_Y) * 512 + (size_t)lane * 16;
    const __bf16* b0r = hT + (size_t)(nb0 + n) * HID + kb;
    const __bf16* b1r = hT + (size_t)(nb1 + n) * HID + kb;

    #pragma unroll 4
    for (int kt = 0; kt < KT_Y; ++kt) {
        __builtin_prefetch(a0 + (size_t)(kt + 8) * 512, 0, 3);
        v16bf af0 = *(const v16bf*)(a0 + (size_t)kt * 512);
        v16bf af1 = *(const v16bf*)(a1 + (size_t)kt * 512);
        v16bf b0  = *(const v16bf*)(b0r + kt * 32);
        v16bf b1  = *(const v16bf*)(b1r + kt * 32);
        c00 = wmma_bf16(af0, b0, c00);
        c01 = wmma_bf16(af0, b1, c01);
        c10 = wmma_bf16(af1, b0, c10);
        c11 = wmma_bf16(af1, b1, c11);
    }

    const int half = (lane < 16) ? 0 : 8;
    const int mb0  = mt0 * 16 + half;
    const int mb1  = mt1 * 16 + half;
    v8f bias0 = *(const v8f*)(by + mb0);
    v8f bias1 = *(const v8f*)(by + mb1);
    #pragma unroll
    for (int r = 0; r < 8; ++r) {
        out[(size_t)(mb0 + r) * BATCH + nb0 + n] =
            1.0f / (1.0f + __expf(-(c00[r] + bias0[r])));
        out[(size_t)(mb0 + r) * BATCH + nb1 + n] =
            1.0f / (1.0f + __expf(-(c01[r] + bias0[r])));
        out[(size_t)(mb1 + r) * BATCH + nb0 + n] =
            1.0f / (1.0f + __expf(-(c10[r] + bias1[r])));
        out[(size_t)(mb1 + r) * BATCH + nb1 + n] =
            1.0f / (1.0f + __expf(-(c11[r] + bias1[r])));
    }
}

// ---------------------------------------------------------------------------
// host side
// ---------------------------------------------------------------------------
extern "C" void kernel_launch(void* const* d_in, const int* in_sizes, int n_in,
                              void* d_out, int out_size, void* d_ws, size_t ws_size,
                              hipStream_t stream) {
    const float* xs  = (const float*)d_in[0];  // [T][IN][BATCH]
    const float* W_h = (const float*)d_in[1];  // [HID][KH]
    const float* b_h = (const float*)d_in[2];  // [HID]
    const float* W_y = (const float*)d_in[3];  // [WID][HID]
    const float* b_y = (const float*)d_in[4];  // [WID]
    const float* h0  = (const float*)d_in[5];  // [HID]
    float* out = (float*)d_out;

    // workspace layout (bytes)
    const size_t WP_BYTES  = (size_t)(HID / 16) * KT_H * 32 * 16 * 2;  // 3 MiB
    const size_t WYP_BYTES = (size_t)(WID / 16) * KT_Y * 32 * 16 * 2;  // 1 MiB
    const size_t H_BYTES   = (size_t)BATCH * HID * 2;                  // 0.5 MiB
    char* ws = (char*)d_ws;
    __bf16* Wp  = (__bf16*)(ws);
    __bf16* Wyp = (__bf16*)(ws + WP_BYTES);
    __bf16* hA  = (__bf16*)(ws + WP_BYTES + WYP_BYTES);
    __bf16* hB  = (__bf16*)(ws + WP_BYTES + WYP_BYTES + H_BYTES);
    __bf16* xT  = (__bf16*)(ws + WP_BYTES + WYP_BYTES + 2 * H_BYTES);

    // one-time packs + h init (re-run every call: deterministic)
    pack_wh<<<(int)(WP_BYTES / 2 / 256), 256, 0, stream>>>(W_h, Wp);
    pack_wy<<<(int)(WYP_BYTES / 2 / 256), 256, 0, stream>>>(W_y, Wyp);
    init_h<<<(BATCH * HID) / 256, 256, 0, stream>>>(h0, hA);

    __bf16* hc = hA;
    __bf16* hn = hB;
    for (int t = 0; t < T_STEPS; ++t) {
        const float* xt = xs + (size_t)t * IN_DIM * BATCH;
        xpose<<<(IN_DIM * BATCH) / 256, 256, 0, stream>>>(xt, xT);
        elman_hstep<<<dim3(8, 8), 128, 0, stream>>>(Wp, hc, xT, b_h, hn);
        elman_ystep<<<dim3(8, 4), 128, 0, stream>>>(Wyp, hn, b_y,
                                                    out + (size_t)t * WID * BATCH);
        __bf16* tmp = hc; hc = hn; hn = tmp;
    }
}
